// GPTReimplementation_72086731096332
// MI455X (gfx1250) — matrix operand order repaired and checked by reading, not compiled, not run on hardware
//
#include <hip/hip_runtime.h>
#include <hip/hip_bf16.h>
#include <math.h>

// ---- model dims ----
#define D_    768
#define F_    3072        // 4*D
#define L_    12
#define V_    50257
#define VPAD_ 50304       // 393 * 128
#define T_    2048
#define M_    4096        // B*T
#define EPS_  1e-5f

typedef __bf16 bf16_t;
typedef __attribute__((ext_vector_type(16))) __bf16 v16bf;
typedef __attribute__((ext_vector_type(8)))  __bf16 v8bf;
typedef __attribute__((ext_vector_type(8)))  float  v8f;

// ---------------- helpers ----------------
__device__ inline v16bf pack16(v8bf lo, v8bf hi) {
  v16bf r;
#pragma unroll
  for (int i = 0; i < 8; ++i) { r[i] = lo[i]; r[i + 8] = hi[i]; }
  return r;
}

__device__ inline float gelu_exact(float v) {
  return 0.5f * v * (1.0f + erff(v * 0.70710678118654752f));
}

// ---------------- 1) convert w_embed f32 -> bf16, zero-pad rows to VPAD_ ----
__global__ __launch_bounds__(256) void cvt_wembed_kernel(const float* __restrict__ src,
                                                         bf16_t* __restrict__ dst) {
  size_t idx = (size_t)blockIdx.x * 256 + threadIdx.x;   // grid covers VPAD_*D_
  float v = (idx < (size_t)V_ * D_) ? src[idx] : 0.0f;
  dst[idx] = (bf16_t)v;
}

// ---------------- 2) embedding gather + sinusoidal pos: x = 2*e + pos -------
__global__ __launch_bounds__(256) void embed_pos_kernel(const int* __restrict__ tok,
                                                        const float* __restrict__ wemb,
                                                        float* __restrict__ x) {
  size_t idx = (size_t)blockIdx.x * 256 + threadIdx.x;   // grid covers M_*D_
  int d   = (int)(idx % D_);
  int row = (int)(idx / D_);
  int t   = row % T_;
  int token = tok[row];
  float e = wemb[(size_t)token * D_ + d];
  int half = d >> 1;
  // (1/10000)^((2/D)*half) = exp(-ln(10000) * 2*half/D)
  float freq = expf(-9.210340371976184f * (2.0f / (float)D_) * (float)half);
  float ang  = (float)t * freq;
  float pos  = (d & 1) ? cosf(ang) : sinf(ang);
  x[idx] = 2.0f * e + pos;
}

// ---------------- 3) transpose+convert: src f32 [K][N] -> dst bf16 [N][K] ---
__global__ __launch_bounds__(256) void transpose_cvt_kernel(const float* __restrict__ src,
                                                            bf16_t* __restrict__ dst,
                                                            int K, int N) {
  __shared__ float tile[32][33];
  int tx = threadIdx.x & 31;
  int ty = threadIdx.x >> 5;          // 0..7
  int nBase = blockIdx.x * 32;
  int kBase = blockIdx.y * 32;
#pragma unroll
  for (int j = 0; j < 32; j += 8)
    tile[ty + j][tx] = src[(size_t)(kBase + ty + j) * N + (nBase + tx)];
  __syncthreads();
#pragma unroll
  for (int j = 0; j < 32; j += 8)
    dst[(size_t)(nBase + ty + j) * K + (kBase + tx)] = (bf16_t)tile[tx][ty + j];
}

// ---------------- 4) buggy layernorm (subtracts unbiased std) -> bf16 rows --
__global__ __launch_bounds__(256) void layernorm_kernel(const float* __restrict__ x,
                                                        const float* __restrict__ scale,
                                                        bf16_t* __restrict__ out) {
  __shared__ float sSum[256];
  __shared__ float sSq[256];
  __shared__ float sStd;
  int row = blockIdx.x;
  int tid = threadIdx.x;
  const float* xr = x + (size_t)row * D_;
  float v0 = xr[tid], v1 = xr[tid + 256], v2 = xr[tid + 512];
  sSum[tid] = v0 + v1 + v2;
  sSq[tid]  = v0 * v0 + v1 * v1 + v2 * v2;
  __syncthreads();
  for (int off = 128; off > 0; off >>= 1) {
    if (tid < off) { sSum[tid] += sSum[tid + off]; sSq[tid] += sSq[tid + off]; }
    __syncthreads();
  }
  if (tid == 0) {
    float sum = sSum[0], sq = sSq[0];
    float mean = sum * (1.0f / (float)D_);
    float var  = (sq - sum * mean) * (1.0f / (float)(D_ - 1));   // unbiased (ddof=1)
    sStd = sqrtf(fmaxf(var, 0.0f));
  }
  __syncthreads();
  float stdv = sStd;
  float inv  = 1.0f / fmaxf(stdv, EPS_);
  bf16_t* orow = out + (size_t)row * D_;
  orow[tid]       = (bf16_t)(scale[tid]       * (v0 - stdv) * inv);
  orow[tid + 256] = (bf16_t)(scale[tid + 256] * (v1 - stdv) * inv);
  orow[tid + 512] = (bf16_t)(scale[tid + 512] * (v2 - stdv) * inv);
}

// ---------------- 5) bf16 WMMA GEMM: C[M][Nreal] = op(A[M][K] * B[N][K]^T) --
// A row-major [m][k]; B packed [n][k] (K contiguous per output column).
// Block tile 128x128, 256 threads = 8 waves; wave tile 32(M) x 64(N).
// K loop unrolled x2 with register ping-pong (requires K % 64 == 0).
template <bool GELU, bool RESID, bool OUTBF16, bool STREAM>
__global__ __launch_bounds__(256) void gemm_wmma_bf16_kernel(
    const bf16_t* __restrict__ A, const bf16_t* __restrict__ B,
    void* __restrict__ Cout, const float* __restrict__ Res,
    int K, int Nreal) {
  const int lane  = threadIdx.x & 31;
  const int wave  = threadIdx.x >> 5;           // 0..7
  const int mBlk  = blockIdx.y * 128;
  const int nBlk  = blockIdx.x * 128;
  const int waveM = (wave & 3) * 32;            // 4 sub-tiles of 32 rows
  const int waveN = (wave >> 2) * 64;           // 2 sub-tiles of 64 cols
  const int l16   = lane & 15;
  const int kSel  = lane >> 4;                  // 0 or 1 (half-wave)

  // Per-lane fragment base pointers (fixed K offsets folded per ISA layout):
  //  A 16x32: half-wave0 VGPR0-3=K[0..7],  VGPR4-7=K[16..23]
  //           half-wave1 VGPR0-3=K[8..15], VGPR4-7=K[24..31]
  //  B 32x16: same K layout as A (K[kSel*8..+7], K[16+kSel*8..+7])
  const bf16_t* aBase[2];
  const bf16_t* bBase[4];
#pragma unroll
  for (int mi = 0; mi < 2; ++mi)
    aBase[mi] = A + (size_t)(mBlk + waveM + mi * 16 + l16) * K + kSel * 8;
#pragma unroll
  for (int ni = 0; ni < 4; ++ni)
    bBase[ni] = B + (size_t)(nBlk + waveN + ni * 16 + l16) * K + kSel * 8;

  v8f zero = {};
  v8f acc[2][4];
#pragma unroll
  for (int mi = 0; mi < 2; ++mi)
#pragma unroll
    for (int ni = 0; ni < 4; ++ni) acc[mi][ni] = zero;

  v16bf a0[2], b0[4], a1[2], b1[4];
  // preload K-tile 0
#pragma unroll
  for (int mi = 0; mi < 2; ++mi)
    a0[mi] = pack16(*reinterpret_cast<const v8bf*>(aBase[mi]),
                    *reinterpret_cast<const v8bf*>(aBase[mi] + 16));
#pragma unroll
  for (int ni = 0; ni < 4; ++ni)
    b0[ni] = pack16(*reinterpret_cast<const v8bf*>(bBase[ni]),
                    *reinterpret_cast<const v8bf*>(bBase[ni] + 16));

  for (int k0 = 0; k0 < K; k0 += 64) {
    // load K-tile (k0+32) into ping buffer -- always in-bounds since K%64==0
#pragma unroll
    for (int mi = 0; mi < 2; ++mi)
      a1[mi] = pack16(*reinterpret_cast<const v8bf*>(aBase[mi] + k0 + 32),
                      *reinterpret_cast<const v8bf*>(aBase[mi] + k0 + 48));
#pragma unroll
    for (int ni = 0; ni < 4; ++ni)
      b1[ni] = pack16(*reinterpret_cast<const v8bf*>(bBase[ni] + k0 + 32),
                      *reinterpret_cast<const v8bf*>(bBase[ni] + k0 + 48));

#pragma unroll
    for (int mi = 0; mi < 2; ++mi)
#pragma unroll
      for (int ni = 0; ni < 4; ++ni)
        acc[mi][ni] = __builtin_amdgcn_wmma_f32_16x16x32_bf16(
            false, a0[mi], false, b0[ni], (short)0, acc[mi][ni], false, false);

    if (k0 + 64 < K) {   // uniform: load K-tile (k0+64) into pong buffer
#pragma unroll
      for (int mi = 0; mi < 2; ++mi)
        a0[mi] = pack16(*reinterpret_cast<const v8bf*>(aBase[mi] + k0 + 64),
                        *reinterpret_cast<const v8bf*>(aBase[mi] + k0 + 80));
#pragma unroll
      for (int ni = 0; ni < 4; ++ni)
        b0[ni] = pack16(*reinterpret_cast<const v8bf*>(bBase[ni] + k0 + 64),
                        *reinterpret_cast<const v8bf*>(bBase[ni] + k0 + 80));
      __builtin_prefetch(aBase[0] + k0 + 128, 0, 3);
      __builtin_prefetch(bBase[0] + k0 + 128, 0, 3);
    }

#pragma unroll
    for (int mi = 0; mi < 2; ++mi)
#pragma unroll
      for (int ni = 0; ni < 4; ++ni)
        acc[mi][ni] = __builtin_amdgcn_wmma_f32_16x16x32_bf16(
            false, a1[mi], false, b1[ni], (short)0, acc[mi][ni], false, false);
  }

  // Epilogue. C/D layout: VGPR r -> row = r + (lane>=16 ? 8 : 0), col = lane&15.
  const int rowOff = kSel * 8;
#pragma unroll
  for (int mi = 0; mi < 2; ++mi) {
#pragma unroll
    for (int ni = 0; ni < 4; ++ni) {
      int col = nBlk + waveN + ni * 16 + l16;
      if (col < Nreal) {
#pragma unroll
        for (int r = 0; r < 8; ++r) {
          int row = mBlk + waveM + mi * 16 + rowOff + r;
          float v = acc[mi][ni][r];
          if (RESID) v += Res[(size_t)row * Nreal + col];
          if (GELU)  v = gelu_exact(v);
          if (OUTBF16) {
            ((bf16_t*)Cout)[(size_t)row * Nreal + col] = (bf16_t)v;
          } else if (STREAM) {
            // 823MB streaming logits: non-temporal so w_embed stays L2-resident
            __builtin_nontemporal_store(v, (float*)Cout + (size_t)row * Nreal + col);
          } else {
            ((float*)Cout)[(size_t)row * Nreal + col] = v;
          }
        }
      }
    }
  }
}

// ---------------- workspace layout (bytes) ----------------
#define OFF_X     ((size_t)0)                                   // f32  [M][D]     12.6 MB
#define OFF_XLN   (OFF_X    + (size_t)M_ * D_ * 4)              // bf16 [M][D]      6.3 MB
#define OFF_ACT   (OFF_XLN  + (size_t)M_ * D_ * 2)              // bf16 [M][F]     25.2 MB
#define OFF_ENCB  (OFF_ACT  + (size_t)M_ * F_ * 2)              // bf16 [F][D]      4.7 MB
#define OFF_PROJB (OFF_ENCB + (size_t)F_ * D_ * 2)              // bf16 [D][F]      4.7 MB
#define OFF_WEMB  (OFF_PROJB+ (size_t)F_ * D_ * 2)              // bf16 [VPAD][D]  77.3 MB

extern "C" void kernel_launch(void* const* d_in, const int* in_sizes, int n_in,
                              void* d_out, int out_size, void* d_ws, size_t ws_size,
                              hipStream_t stream) {
  const int*   tok     = (const int*)d_in[0];
  const float* w_embed = (const float*)d_in[1];
  const float* ln2     = (const float*)d_in[2];
  const float* enc_w   = (const float*)d_in[3];
  const float* proj_w  = (const float*)d_in[4];
  const float* fscale  = (const float*)d_in[5];
  float* out = (float*)d_out;

  char* ws = (char*)d_ws;
  float*  x     = (float*)(ws + OFF_X);
  bf16_t* xln   = (bf16_t*)(ws + OFF_XLN);
  bf16_t* act   = (bf16_t*)(ws + OFF_ACT);
  bf16_t* encb  = (bf16_t*)(ws + OFF_ENCB);
  bf16_t* projb = (bf16_t*)(ws + OFF_PROJB);
  bf16_t* wembb = (bf16_t*)(ws + OFF_WEMB);

  // w_embed [V][D] is ALREADY in packed B layout [n][k] for the logits GEMM.
  cvt_wembed_kernel<<<(VPAD_ * D_) / 256, 256, 0, stream>>>(w_embed, wembb);

  // x = 2*embed + sinusoidal pos
  embed_pos_kernel<<<(M_ * (size_t)D_) / 256, 256, 0, stream>>>(tok, w_embed, x);

  for (int l = 0; l < L_; ++l) {
    // enc_w[l]: f32 [D][F] -> bf16 [F][D] ; proj_w[l]: f32 [F][D] -> bf16 [D][F]
    transpose_cvt_kernel<<<dim3(F_ / 32, D_ / 32), 256, 0, stream>>>(
        enc_w + (size_t)l * D_ * F_, encb, D_, F_);
    transpose_cvt_kernel<<<dim3(D_ / 32, F_ / 32), 256, 0, stream>>>(
        proj_w + (size_t)l * F_ * D_, projb, F_, D_);

    layernorm_kernel<<<M_, 256, 0, stream>>>(x, ln2 + (size_t)l * D_, xln);

    // h = gelu(ln(x) @ enc_w) : [4096,768]x[768,3072] -> bf16
    gemm_wmma_bf16_kernel<true, false, true, false>
        <<<dim3(F_ / 128, M_ / 128), 256, 0, stream>>>(xln, encb, act, nullptr, D_, F_);

    // x = x + h @ proj_w : [4096,3072]x[3072,768] -> f32 (+residual)
    gemm_wmma_bf16_kernel<false, true, false, false>
        <<<dim3(D_ / 128, M_ / 128), 256, 0, stream>>>(act, projb, x, x, F_, D_);
  }

  // final LN + logits = ln(x) @ w_embed^T : [4096,768]x[768,50257]
  layernorm_kernel<<<M_, 256, 0, stream>>>(x, fscale, xln);
  gemm_wmma_bf16_kernel<false, false, false, true>
      <<<dim3(VPAD_ / 128, M_ / 128), 256, 0, stream>>>(xln, wembb, out, nullptr, D_, V_);
}
